// Grid_Embedding_38062000177905
// MI455X (gfx1250) — matrix operand order repaired and checked by reading, not compiled, not run on hardware
//
#include <hip/hip_runtime.h>

// ---------------------------------------------------------------------------
// MI455X (gfx1250) implementation: bandwidth-bound problem (~134MB min traffic
// @23.3TB/s ~ 5.8us; only 21.4 GFLOP), so stream X once, keep intermediates in
// bf16, and run all GEMMs on v_wmma_f32_16x16x32_bf16 with f32 accumulation.
// ---------------------------------------------------------------------------

typedef __attribute__((ext_vector_type(16))) __bf16 v16bf;
typedef __attribute__((ext_vector_type(8)))  __bf16 v8bf;
typedef __attribute__((ext_vector_type(8)))  float  v8f;

union AF { v16bf v; v8bf h[2]; };

__device__ __forceinline__ v8f wmma_bf16(AF a, AF b, v8f c) {
  // (neg_a, A, neg_b, B, c_mod, C, reuse_a, reuse_b)
  return __builtin_amdgcn_wmma_f32_16x16x32_bf16(false, a.v, false, b.v,
                                                 (short)0, c, false, false);
}

constexpr int O  = 256;   // grid cells (== D of X last two dims)
constexpr int DM = 128;   // d_model
constexpr int BL = 256;   // B*L tiles

// LDS pitches (bf16 elements); all row strides are multiples of 16B
constexpr int XS_P = 264;
constexpr int YS_P = 136;
constexpr int YT_P = 264;

// ---- workspace layout (bytes) ----
constexpr size_t OFF_W1T  = 0;                      // [128][512] bf16
constexpr size_t OFF_W2T  = OFF_W1T + 128*512*2;    // [128][128] bf16
constexpr size_t OFF_W3T  = OFF_W2T + 128*128*2;    // [128][128] bf16
constexpr size_t OFF_DISW = OFF_W3T + 128*128*2;    // [256][256] bf16
constexpr size_t OFF_DEGN = OFF_DISW + 256*256*2;   // [BL][256]  bf16
constexpr size_t OFF_MASK = OFF_DEGN + (size_t)BL*256*2;    // [BL][256][8] u32
constexpr size_t OFF_Y    = OFF_MASK + (size_t)BL*256*8*4;  // [BL][256][128] bf16

// ---- stage 1 LDS layout ----
constexpr int S1_XS   = 0;                 // bf16 [256][XS_P]
constexpr int S1_XST  = 256*XS_P*2;        // bf16 [256][XS_P]
constexpr int S1_RED  = S1_XST + 256*XS_P*2;   // float [256] reduction scratch
constexpr int S1_SIZE = S1_RED + 256*4;        // ~271.4 KB (<=320KB WGP LDS)

// ---- stage 2 LDS layout ----
constexpr int S2_YS   = 0;                 // bf16 [256][YS_P]
constexpr int S2_YST  = S2_YS  + 256*YS_P*2;   // bf16 [128][YT_P]
constexpr int S2_ZG   = S2_YST + 128*YT_P*2;   // bf16 [256][YS_P]
constexpr int S2_ZS   = S2_ZG  + 256*YS_P*2;   // bf16 [256][YS_P]
constexpr int S2_MASK = S2_ZS  + 256*YS_P*2;   // u32  [256][8]
constexpr int S2_DEGN = S2_MASK + 256*8*4;     // bf16 [256]
constexpr int S2_SIZE = S2_DEGN + 256*2;       // ~285.2 KB

// ===========================================================================
// Prep: transposed bf16 weights (B operand wants column-major contiguous K)
// ===========================================================================
__global__ void prep_weights(const float* __restrict__ W1,
                             const float* __restrict__ W2,
                             const float* __restrict__ W3,
                             __bf16* __restrict__ W1T,
                             __bf16* __restrict__ W2T,
                             __bf16* __restrict__ W3T) {
  int tid = blockIdx.x * blockDim.x + threadIdx.x;
  if (tid < 512 * 128) {          // W1 [512][128] -> W1T [128][512]
    int n = tid & 127, k = tid >> 7;
    W1T[n * 512 + k] = (__bf16)W1[tid];
  }
  if (tid < 128 * 128) {          // W2/W3 [128][128] -> [128][128]^T
    int n = tid & 127, k = tid >> 7;
    W2T[n * 128 + k] = (__bf16)W2[tid];
    W3T[n * 128 + k] = (__bf16)W3[tid];
  }
}

// dis_w[i][j] = (dis<=2 ? sqrt(dis) : 0) / sum_j sqrt(dis[i][j])
__global__ void prep_disw(const float* __restrict__ dis, __bf16* __restrict__ disw) {
  int i = threadIdx.x;                 // 256 threads, 1 block
  float rs = 0.f;
  for (int j = 0; j < O; ++j) rs += sqrtf(dis[i * O + j]);
  float inv = 1.f / rs;
  for (int j = 0; j < O; ++j) {
    float d = dis[i * O + j];
    float w = (d <= 2.0f) ? sqrtf(d) * inv : 0.f;
    disw[i * O + j] = (__bf16)w;
  }
}

// ===========================================================================
// Stage 1: per (b,l) tile — Y = [X | X^T] @ W1 + b1 (bf16 out),
//          tile_deg / sum_deg / packed sem mask.
// ===========================================================================
__global__ void __launch_bounds__(256)
stage1(const float* __restrict__ X, const float* __restrict__ b1,
       const __bf16* __restrict__ W1T,
       __bf16* __restrict__ Yw, __bf16* __restrict__ degnW,
       unsigned* __restrict__ maskW) {
  extern __shared__ char smem[];
  __bf16* Xs  = (__bf16*)(smem + S1_XS);
  __bf16* XsT = (__bf16*)(smem + S1_XST);
  float*  red = (float*)(smem + S1_RED);

  const int tile = blockIdx.x;
  const int t    = threadIdx.x;
  const float* Xt = X + (size_t)tile * O * O;

  // ---- load X tile, f32 -> bf16, row-major + transposed copies ----
  for (int it = 0; it < 64; ++it) {
    int linear = it * 1024 + t * 4;
    int row = linear >> 8, col = linear & 255;
    float4 x4 = *(const float4*)(Xt + linear);
    __bf16 e0 = (__bf16)x4.x, e1 = (__bf16)x4.y, e2 = (__bf16)x4.z, e3 = (__bf16)x4.w;
    __bf16* dst = Xs + row * XS_P + col;
    dst[0] = e0; dst[1] = e1; dst[2] = e2; dst[3] = e3;
    XsT[(col + 0) * XS_P + row] = e0;
    XsT[(col + 1) * XS_P + row] = e1;
    XsT[(col + 2) * XS_P + row] = e2;
    XsT[(col + 3) * XS_P + row] = e3;
  }
  __syncthreads();

  // ---- tile_deg[j] (vectorized b128 LDS reads), packed mask row j ----
  float td = 0.f;
  {
    const int j = t;
    const __bf16* r0 = Xs  + j * XS_P;
    const __bf16* r1 = XsT + j * XS_P;
    unsigned* mdst = maskW + ((size_t)tile * O + j) * 8;
#pragma unroll
    for (int w = 0; w < 8; ++w) {           // 8 mask words of 32 bits
      unsigned mw = 0u;
#pragma unroll
      for (int k8 = 0; k8 < 32; k8 += 8) {
        v8bf a8 = *(const v8bf*)(r0 + w * 32 + k8);
        v8bf c8 = *(const v8bf*)(r1 + w * 32 + k8);
#pragma unroll
        for (int e = 0; e < 8; ++e) {
          float a = (float)a8[e], c = (float)c8[e];
          td += a + c;
          if (a > 0.f || c > 0.f) mw |= (1u << (k8 + e));
        }
      }
      mdst[w] = mw;
    }
  }

  // ---- deterministic fixed-order tree reduction for sum_deg ----
  red[t] = td;
  __syncthreads();
#pragma unroll
  for (int s = 128; s > 0; s >>= 1) {
    if (t < s) red[t] += red[t + s];
    __syncthreads();
  }
  const float sum_deg = red[0];
  degnW[(size_t)tile * O + t] = (__bf16)(td / sum_deg);

  // ---- Y = [Xs | XsT] @ W1  (K = 512) ----
  const int lane = t & 31;
  const int wave = t >> 5;
  const int half = lane >> 4;
  const int l15  = lane & 15;

  for (int mt2 = 0; mt2 < 2; ++mt2) {
    const int mt   = wave * 2 + mt2;       // M tile 0..15
    const int mrow = mt * 16 + l15;        // A row for this lane
    v8f acc[8] = {};
    for (int kt = 0; kt < 16; ++kt) {
      const int k0 = kt * 32;
      const __bf16* arow = (k0 < 256) ? (Xs  + mrow * XS_P + k0)
                                      : (XsT + mrow * XS_P + (k0 - 256));
      AF a;
      a.h[0] = *(const v8bf*)(arow + 8 * half);
      a.h[1] = *(const v8bf*)(arow + 16 + 8 * half);
#pragma unroll
      for (int nt = 0; nt < 8; ++nt) {
        const int n = nt * 16 + l15;
        const __bf16* bp = W1T + n * 512 + k0 + 8 * half;
        AF bfr;
        bfr.h[0] = *(const v8bf*)bp;
        bfr.h[1] = *(const v8bf*)(bp + 16);
        acc[nt] = wmma_bf16(a, bfr, acc[nt]);
      }
    }
    // store Y (bf16) + bias
    __bf16* ytile = Yw + (size_t)tile * (O * DM);
#pragma unroll
    for (int nt = 0; nt < 8; ++nt) {
      const int col = nt * 16 + l15;
      const float bias = b1[col];
      const int rbase = mt * 16 + half * 8;
#pragma unroll
      for (int v = 0; v < 8; ++v)
        ytile[(size_t)(rbase + v) * DM + col] = (__bf16)(acc[nt][v] + bias);
    }
  }
}

// ===========================================================================
// Stage 2: per tile — geo/sem aggregation GEMMs (shared B fragments),
//          Z = Y + agg (via LDS re-layout), output projections, fp32 store.
// ===========================================================================
__global__ void __launch_bounds__(256)
stage2(const __bf16* __restrict__ Yw, const __bf16* __restrict__ disw,
       const __bf16* __restrict__ degnW, const unsigned* __restrict__ maskW,
       const __bf16* __restrict__ W2T, const __bf16* __restrict__ W3T,
       const float* __restrict__ b2, const float* __restrict__ b3,
       float* __restrict__ out) {
  extern __shared__ char smem[];
  __bf16*  Ys    = (__bf16*)(smem + S2_YS);    // [256][YS_P]
  __bf16*  YsT   = (__bf16*)(smem + S2_YST);   // [128][YT_P]
  __bf16*  Zg    = (__bf16*)(smem + S2_ZG);    // [256][YS_P]
  __bf16*  Zs    = (__bf16*)(smem + S2_ZS);    // [256][YS_P]
  unsigned* maskS = (unsigned*)(smem + S2_MASK);
  __bf16*  degnS = (__bf16*)(smem + S2_DEGN);

  const int tile = blockIdx.x;
  const int t    = threadIdx.x;
  const __bf16* Yt = Yw + (size_t)tile * (O * DM);

  // ---- stage Y (row-major + transposed), mask, degn into LDS ----
  for (int it = 0; it < 16; ++it) {
    int linear = it * 2048 + t * 8;
    int row = linear >> 7, col = linear & 127;
    v8bf y8 = *(const v8bf*)(Yt + linear);
    *(v8bf*)(Ys + row * YS_P + col) = y8;
#pragma unroll
    for (int e = 0; e < 8; ++e) YsT[(col + e) * YT_P + row] = y8[e];
  }
  {
    const unsigned* msrc = maskW + ((size_t)tile * O + t) * 8;
#pragma unroll
    for (int w = 0; w < 8; ++w) maskS[t * 8 + w] = msrc[w];
    degnS[t] = degnW[(size_t)tile * O + t];
  }
  __syncthreads();

  const int lane = t & 31;
  const int wave = t >> 5;
  const int half = lane >> 4;
  const int l15  = lane & 15;

  for (int mt2 = 0; mt2 < 2; ++mt2) {
    const int mt   = wave * 2 + mt2;
    const int mrow = mt * 16 + l15;
    const unsigned* mrow_mask = maskS + mrow * 8;

    // ---- agg GEMMs: geo & sem share the Y (B) fragments, K = 256 ----
    v8f accG[8] = {}, accS[8] = {};
    for (int kt = 0; kt < 8; ++kt) {
      const int k0 = kt * 32;
      AF ag;                                   // A_geo = dis_w rows (L2 hot)
      const __bf16* gp = disw + mrow * O + k0 + 8 * half;
      ag.h[0] = *(const v8bf*)gp;
      ag.h[1] = *(const v8bf*)(gp + 16);
      AF as;                                   // A_sem = mask ? degn[k] : 0
      const unsigned mword = mrow_mask[kt];
#pragma unroll
      for (int e = 0; e < 8; ++e) {
        const int kk0 = 8 * half + e;
        const int kk1 = 16 + 8 * half + e;
        as.h[0][e] = ((mword >> kk0) & 1u) ? degnS[k0 + kk0] : (__bf16)0.f;
        as.h[1][e] = ((mword >> kk1) & 1u) ? degnS[k0 + kk1] : (__bf16)0.f;
      }
#pragma unroll
      for (int nt = 0; nt < 8; ++nt) {
        const int n = nt * 16 + l15;
        const __bf16* bp = YsT + n * YT_P + k0 + 8 * half;
        AF bfr;
        bfr.h[0] = *(const v8bf*)bp;
        bfr.h[1] = *(const v8bf*)(bp + 16);
        accG[nt] = wmma_bf16(ag, bfr, accG[nt]);
        accS[nt] = wmma_bf16(as, bfr, accS[nt]);
      }
    }

    // ---- Z = Y + agg, re-layout C->A through LDS (wave-private rows) ----
#pragma unroll
    for (int nt = 0; nt < 8; ++nt) {
      const int col = nt * 16 + l15;
      const int rbase = mt * 16 + half * 8;
#pragma unroll
      for (int v = 0; v < 8; ++v) {
        const int row = rbase + v;
        const float y = (float)Ys[row * YS_P + col];
        Zg[row * YS_P + col] = (__bf16)(y + accG[nt][v]);
        Zs[row * YS_P + col] = (__bf16)(y + accS[nt][v]);
      }
    }
    // same wave wrote these rows; LDS ops are in-order per wave -> no barrier

    // ---- output projections, K = 128 ----
    v8f og[8] = {}, os_[8] = {};
    for (int kt = 0; kt < 4; ++kt) {
      const int k0 = kt * 32;
      AF azg, azs;
      const __bf16* zgp = Zg + mrow * YS_P + k0;
      azg.h[0] = *(const v8bf*)(zgp + 8 * half);
      azg.h[1] = *(const v8bf*)(zgp + 16 + 8 * half);
      const __bf16* zsp = Zs + mrow * YS_P + k0;
      azs.h[0] = *(const v8bf*)(zsp + 8 * half);
      azs.h[1] = *(const v8bf*)(zsp + 16 + 8 * half);
#pragma unroll
      for (int nt = 0; nt < 8; ++nt) {
        const int n = nt * 16 + l15;
        const __bf16* p2 = W2T + n * DM + k0 + 8 * half;
        const __bf16* p3 = W3T + n * DM + k0 + 8 * half;
        AF b2f, b3f;
        b2f.h[0] = *(const v8bf*)p2;  b2f.h[1] = *(const v8bf*)(p2 + 16);
        b3f.h[0] = *(const v8bf*)p3;  b3f.h[1] = *(const v8bf*)(p3 + 16);
        og[nt]  = wmma_bf16(azg, b2f, og[nt]);
        os_[nt] = wmma_bf16(azs, b3f, os_[nt]);
      }
    }

    // ---- store out[tile][row][0:128]=geo, [128:256]=sem (fp32) ----
    float* obase = out + (size_t)tile * O * 256;
#pragma unroll
    for (int nt = 0; nt < 8; ++nt) {
      const int col = nt * 16 + l15;
      const float bg = b2[col], bs = b3[col];
      const int rbase = mt * 16 + half * 8;
#pragma unroll
      for (int v = 0; v < 8; ++v) {
        const int row = rbase + v;
        obase[row * 256 + col]       = og[nt][v] + bg;
        obase[row * 256 + 128 + col] = os_[nt][v] + bs;
      }
    }
  }
}

// ===========================================================================
extern "C" void kernel_launch(void* const* d_in, const int* in_sizes, int n_in,
                              void* d_out, int out_size, void* d_ws, size_t ws_size,
                              hipStream_t stream) {
  const float* X   = (const float*)d_in[0];
  const float* dis = (const float*)d_in[1];
  const float* W1  = (const float*)d_in[2];
  const float* b1  = (const float*)d_in[3];
  const float* W2  = (const float*)d_in[4];
  const float* b2  = (const float*)d_in[5];
  const float* W3  = (const float*)d_in[6];
  const float* b3  = (const float*)d_in[7];
  float* out = (float*)d_out;

  char* ws = (char*)d_ws;
  __bf16*   W1T   = (__bf16*)(ws + OFF_W1T);
  __bf16*   W2T   = (__bf16*)(ws + OFF_W2T);
  __bf16*   W3T   = (__bf16*)(ws + OFF_W3T);
  __bf16*   diswB = (__bf16*)(ws + OFF_DISW);
  __bf16*   degnW = (__bf16*)(ws + OFF_DEGN);
  unsigned* maskW = (unsigned*)(ws + OFF_MASK);
  __bf16*   Yw    = (__bf16*)(ws + OFF_Y);

  hipFuncSetAttribute((const void*)stage1, hipFuncAttributeMaxDynamicSharedMemorySize, S1_SIZE);
  hipFuncSetAttribute((const void*)stage2, hipFuncAttributeMaxDynamicSharedMemorySize, S2_SIZE);

  prep_weights<<<256, 256, 0, stream>>>(W1, W2, W3, W1T, W2T, W3T);
  prep_disw<<<1, 256, 0, stream>>>(dis, diswB);
  stage1<<<BL, 256, S1_SIZE, stream>>>(X, b1, W1T, Yw, degnW, maskW);
  stage2<<<BL, 256, S2_SIZE, stream>>>(Yw, diswB, degnW, maskW, W2T, W3T, b2, b3, out);
}
